// FP8DynamicMatmul_45784351375676
// MI455X (gfx1250) — 8-bit matrix operand order repaired and checked by reading, compiled, not run on hardware
//
#include <hip/hip_runtime.h>
#include <hip/hip_bf16.h>


typedef __attribute__((ext_vector_type(16))) int   v16i;
typedef __attribute__((ext_vector_type(8)))  float v8f;

#define NSZ 4096
#define E4M3_AMAX 216.0f


__device__ __forceinline__ unsigned sw_e4m3_1(float x) {
  unsigned u = __float_as_uint(x);
  unsigned s = (u >> 24) & 0x80u;
  float ax = fabsf(x);
  if (ax != ax) return s | 0x7Fu;
  if (ax >= 448.0f) return s | 0x7Eu;
  if (ax < 0.015625f) {
    int q = (int)rintf(ax * 512.0f);
    if (q > 7) q = 7;
    return s | (unsigned)q;
  }
  int e; float m = frexpf(ax, &e);
  int q = (int)rintf(m * 16.0f);
  if (q == 16) { q = 8; e += 1; }
  int ef = e - 1 + 7;
  if (ef <= 0) {
    int qq = (int)rintf(ax * 512.0f);
    if (qq > 7) qq = 7;
    return s | (unsigned)qq;
  }
  return s | ((unsigned)ef << 3) | (unsigned)(q - 8);
}

template <bool HI>
__device__ __forceinline__ int pack2_fp8(float a, float b, int old) {
#if __has_builtin(__builtin_amdgcn_cvt_pk_fp8_f32)
  return __builtin_amdgcn_cvt_pk_fp8_f32(a, b, old, HI);
#else
  unsigned p = sw_e4m3_1(a) | (sw_e4m3_1(b) << 8);
  return HI ? (int)(((unsigned)old & 0x0000FFFFu) | (p << 16))
            : (int)(((unsigned)old & 0xFFFF0000u) | p);
#endif
}

__device__ __forceinline__ unsigned char cvt1_fp8(float a) {
  return (unsigned char)(pack2_fp8<false>(a, a, 0) & 0xFF);
}


__global__ void k_init(unsigned* amax) {
  if (threadIdx.x < 2) amax[threadIdx.x] = 0u;
}


__global__ __launch_bounds__(256) void k_absmax(const float* __restrict__ X,
                                                unsigned* __restrict__ out,
                                                long long n4) {
  const float4* X4 = (const float4*)X;
  float m = 0.0f;
  for (long long i = (long long)blockIdx.x * blockDim.x + threadIdx.x;
       i < n4; i += (long long)gridDim.x * blockDim.x) {
    float4 v = X4[i];
    m = fmaxf(m, fmaxf(fmaxf(fabsf(v.x), fabsf(v.y)),
                       fmaxf(fabsf(v.z), fabsf(v.w))));
  }
  __shared__ float red[256];
  red[threadIdx.x] = m;
  __syncthreads();
  for (int s = 128; s > 0; s >>= 1) {
    if ((int)threadIdx.x < s)
      red[threadIdx.x] = fmaxf(red[threadIdx.x], red[threadIdx.x + s]);
    __syncthreads();
  }
  if (threadIdx.x == 0)
    atomicMax(out, __float_as_uint(red[0]));
}


__global__ __launch_bounds__(256) void k_quantA(const float* __restrict__ X,
                                                unsigned char* __restrict__ Q,
                                                const float* __restrict__ amax) {
  float scale = E4M3_AMAX / amax[0];
  long long i = ((long long)blockIdx.x * blockDim.x + threadIdx.x) * 8;
  float4 x0 = *(const float4*)(X + i);
  float4 x1 = *(const float4*)(X + i + 4);
  int lo = pack2_fp8<false>(x0.x * scale, x0.y * scale, 0);
  lo     = pack2_fp8<true >(x0.z * scale, x0.w * scale, lo);
  int hi = pack2_fp8<false>(x1.x * scale, x1.y * scale, 0);
  hi     = pack2_fp8<true >(x1.z * scale, x1.w * scale, hi);
  int2 w; w.x = lo; w.y = hi;
  *(int2*)(Q + i) = w;
}


#define BT 64
__global__ __launch_bounds__(256) void k_quantBT(const float* __restrict__ X,
                                                 unsigned char* __restrict__ Q,
                                                 const float* __restrict__ amax) {
  __shared__ unsigned char lds[BT][BT + 4];
  float scale = E4M3_AMAX / amax[1];
  int k0 = blockIdx.y * BT;
  int n0 = blockIdx.x * BT;
  int t  = threadIdx.x;
  int tn  = t & 63;
  int tkb = t >> 6;
#pragma unroll
  for (int r = 0; r < 16; ++r) {
    int kk = tkb + 4 * r;
    float v = X[(long long)(k0 + kk) * NSZ + (n0 + tn)];
    lds[tn][kk] = cvt1_fp8(v * scale);
  }
  __syncthreads();
  int tk  = (t & 15) * 4;
  int tnb = t >> 4;
#pragma unroll
  for (int r = 0; r < 4; ++r) {
    int nn = tnb + 16 * r;
    unsigned w = *(const unsigned*)&lds[nn][tk];
    *(unsigned*)(Q + (long long)(n0 + nn) * NSZ + (k0 + tk)) = w;
  }
}


__global__ __launch_bounds__(256) void k_gemm(const unsigned char* __restrict__ A8,
                                              const unsigned char* __restrict__ B8t,
                                              const float* __restrict__ amax,
                                              float* __restrict__ C) {
  const int lane  = threadIdx.x & 31;
  const int wave  = threadIdx.x >> 5;
  const int wm    = wave & 3;
  const int wn    = wave >> 2;
  const int m0    = blockIdx.y * 128 + wm * 32;
  const int n0    = blockIdx.x * 128 + wn * 64;
  const int l16   = lane & 15;
  const int lhalf = lane >> 4;

  v8f acc[2][4];
#pragma unroll
  for (int i = 0; i < 2; ++i)
#pragma unroll
    for (int j = 0; j < 4; ++j) acc[i][j] = (v8f){0,0,0,0,0,0,0,0};

  const unsigned char* aBase[2];
#pragma unroll
  for (int i = 0; i < 2; ++i)
    aBase[i] = A8 + (long long)(m0 + 16 * i + l16) * NSZ + (lhalf ? 16 : 0);
  const unsigned char* bBase[4];
#pragma unroll
  for (int j = 0; j < 4; ++j)
    bBase[j] = B8t + (long long)(n0 + 16 * j + l16) * NSZ + (lhalf ? 16 : 0);

  for (int k0 = 0; k0 < NSZ; k0 += 128) {
    union { v16i v; int2 d[8]; } a[2];
    union { v16i v; int4 q[4]; } b[4];
#pragma unroll
    for (int i = 0; i < 2; ++i) {
      const unsigned char* p = aBase[i] + k0;
#pragma unroll
      for (int j = 0; j < 8; ++j)
        a[i].d[j] = *(const int2*)(p + 32 * (j >> 1) + 8 * (j & 1));
    }
#pragma unroll
    for (int j = 0; j < 4; ++j) {
      const unsigned char* p = bBase[j] + k0;
#pragma unroll
      for (int c = 0; c < 4; ++c)
        b[j].q[c] = *(const int4*)(p + 32 * c);
    }
#pragma unroll
    for (int i = 0; i < 2; ++i)
#pragma unroll
      for (int j = 0; j < 4; ++j)
        acc[i][j] = __builtin_amdgcn_wmma_f32_16x16x128_fp8_fp8(
            a[i].v, b[j].v, (short)0, acc[i][j],
             false,  false);
  }

  const float dq = (amax[0] * amax[1]) * (1.0f / (E4M3_AMAX * E4M3_AMAX));

#pragma unroll
  for (int i = 0; i < 2; ++i) {
#pragma unroll
    for (int j = 0; j < 4; ++j) {
      int col = n0 + 16 * j + l16;
#pragma unroll
      for (int r = 0; r < 8; ++r) {
        int row = m0 + 16 * i + r + lhalf * 8;
        C[(long long)row * NSZ + col] = acc[i][j][r] * dq;
      }
    }
  }
}


extern "C" void kernel_launch(void* const* d_in, const int* in_sizes, int n_in,
                              void* d_out, int out_size, void* d_ws, size_t ws_size,
                              hipStream_t stream) {
  const float* A = (const float*)d_in[0];
  const float* B = (const float*)d_in[1];
  float* C = (float*)d_out;

  unsigned char* ws = (unsigned char*)d_ws;
  unsigned* amaxU = (unsigned*)ws;
  const float* amaxF = (const float*)ws;
  unsigned char* A8  = ws + 256;
  unsigned char* B8t = ws + 256 + (size_t)NSZ * NSZ;

  const long long n  = (long long)NSZ * NSZ;
  const long long n4 = n / 4;

  k_init<<<1, 32, 0, stream>>>(amaxU);
  k_absmax<<<1024, 256, 0, stream>>>(A, amaxU + 0, n4);
  k_absmax<<<1024, 256, 0, stream>>>(B, amaxU + 1, n4);
  k_quantA<<<(unsigned)(n / 8 / 256), 256, 0, stream>>>(A, A8, amaxF);
  k_quantBT<<<dim3(NSZ / BT, NSZ / BT), 256, 0, stream>>>(B, B8t, amaxF);
  k_gemm<<<dim3(NSZ / 128, NSZ / 128), 256, 0, stream>>>(A8, B8t, amaxF, C);
}
